// SymplecticInvariantGNN_40226663694480
// MI455X (gfx1250) — compile-verified, matrix operand order checked by reading, not run on hardware
//
#include <hip/hip_runtime.h>
#include <math.h>

// ---------------------------------------------------------------------------
// CDNA5 (gfx1250) fused Hamiltonian-GNN fwd+bwd using v_wmma_f32_16x16x32_f16
// ---------------------------------------------------------------------------

typedef __attribute__((ext_vector_type(16))) _Float16 v16h;
typedef __attribute__((ext_vector_type(8)))  _Float16 v8h;
typedef __attribute__((ext_vector_type(8)))  float    v8f;

#define NNODES 1024
// swizzled-weight tile offsets (in 512-half tiles)
#define T_We1   0
#define T_We2   8
#define T_We3   40
#define T_We2T  48
#define T_We1T  80
#define T_Wn1   84
#define T_Wn2   92
#define T_Wn3   124
#define T_Wr1   128
#define T_Wr2   136
#define T_Wr2T  168
#define T_Wr1T  200
#define T_Wn3T  204
#define T_Wn2T  212
#define T_Wn1T  244
#define N_TILES 252
#define WS_MSG_BYTES  262144u            // 252*512*2 = 258048, rounded up
#define WS_DM_BYTES   (262144u + 131072u)
#define OUT_DQ 1
#define OUT_DP 3073

__device__ __forceinline__ float fast_rcp(float x) { return __builtin_amdgcn_rcpf(x); }
__device__ __forceinline__ float fast_sqrt(float x) { return __builtin_amdgcn_sqrtf(x); }
// sigmoid via v_exp_f32 + v_rcp_f32 (both TRANS ops, co-execute with WMMA)
__device__ __forceinline__ float fast_sig(float z) {
  return fast_rcp(1.f + __expf(-z));
}

// ---- WMMA fragment helpers -------------------------------------------------
// A fragment (16x32 f16, M x K) from row-major f16 LDS: lane l holds row M=l%16;
// K elements come as two contiguous 8-half chunks at kbase+half*8 and +16 more.
__device__ __forceinline__ v16h load_fragA(const _Float16* lds, int lda, int mbase, int kbase) {
  const int l = threadIdx.x & 31;
  const int row = mbase + (l & 15);
  const int half = l >> 4;
  const _Float16* p0 = lds + row * lda + kbase + half * 8;
  v8h lo = *(const v8h*)(p0);
  v8h hi = *(const v8h*)(p0 + 16);
  v16h r;
#pragma unroll
  for (int e = 0; e < 8; ++e) { r[e] = lo[e]; r[e + 8] = hi[e]; }
  return r;
}

// B fragment from pre-swizzled global weights: lane's 16 halves are contiguous.
__device__ __forceinline__ v16h load_fragB(const _Float16* swz, int tile) {
  const int l = threadIdx.x & 31;
  return *(const v16h*)(swz + (tile * 32 + l) * 16);
}

template <int KTC>
__device__ __forceinline__ v8f wmma_mm(const _Float16* A, int lda, int mt,
                                       const _Float16* Bswz, int ntc, int nt) {
  v8f acc;
#pragma unroll
  for (int r = 0; r < 8; ++r) acc[r] = 0.f;
#pragma unroll
  for (int kt = 0; kt < KTC; ++kt) {
    v16h a = load_fragA(A, lda, mt * 16, kt * 32);
    v16h b = load_fragB(Bswz, kt * ntc + nt);
    acc = __builtin_amdgcn_wmma_f32_16x16x32_f16(false, a, false, b, (short)0, acc,
                                                 false, false);
  }
  return acc;
}

// ---- Epilogues (C/D layout: lane l -> col=l%16, rows (l/16)*8 + r) ---------
template <bool WANT_SP>
__device__ __forceinline__ void epi_silu(v8f acc, const float* __restrict__ bias,
                                         _Float16* out, _Float16* sp, int ldo,
                                         int mt, int nt) {
  const int l = threadIdx.x & 31;
  const int col = nt * 16 + (l & 15);
  const int row0 = mt * 16 + ((l >> 4) << 3);
  const float b = bias[col];
#pragma unroll
  for (int r = 0; r < 8; ++r) {
    float z = acc[r] + b;
    float s = fast_sig(z);
    out[(row0 + r) * ldo + col] = (_Float16)(z * s);
    if (WANT_SP) sp[(row0 + r) * ldo + col] = (_Float16)(s + z * s * (1.f - s));
  }
}

__device__ __forceinline__ void epi_lin_f16(v8f acc, const float* __restrict__ bias,
                                            _Float16* out, int ldo, int mt, int nt) {
  const int l = threadIdx.x & 31;
  const int col = nt * 16 + (l & 15);
  const int row0 = mt * 16 + ((l >> 4) << 3);
  const float b = bias ? bias[col] : 0.f;
#pragma unroll
  for (int r = 0; r < 8; ++r) out[(row0 + r) * ldo + col] = (_Float16)(acc[r] + b);
}

__device__ __forceinline__ void epi_mulsp(v8f acc, const _Float16* sp, _Float16* out,
                                          int ldo, int mt, int nt) {
  const int l = threadIdx.x & 31;
  const int col = nt * 16 + (l & 15);
  const int row0 = mt * 16 + ((l >> 4) << 3);
#pragma unroll
  for (int r = 0; r < 8; ++r) {
    int off = (row0 + r) * ldo + col;
    out[off] = (_Float16)(acc[r] * (float)sp[off]);
  }
}

// ---- Kernel 0: weight swizzle (+ zero d_out) -------------------------------
struct Job { const float* src; int Kdim, Ndim, ntc, trans, tileOff; };
struct JobTab { Job j[15]; };

__global__ __launch_bounds__(256) void prep_kernel(JobTab jt, _Float16* __restrict__ swz,
                                                   float* __restrict__ out, int out_n) {
  const int b = blockIdx.x;
  if (b == 15) {
    for (int k = threadIdx.x; k < out_n; k += 256) out[k] = 0.f;
    return;
  }
  Job jb = jt.j[b];
  const int ktc = (jb.Kdim + 31) >> 5;
  const int total = ktc * jb.ntc * 512;
  _Float16* dst = swz + jb.tileOff * 512;
  for (int idx = threadIdx.x; idx < total; idx += 256) {
    int tile = idx >> 9, w = idx & 511, lane = w >> 4, e = w & 15;
    int kt = tile / jb.ntc, nt = tile % jb.ntc;
    int K = kt * 32 + ((e >> 3) << 4) + ((lane >> 4) << 3) + (e & 7);
    int Nc = nt * 16 + (lane & 15);
    float v = 0.f;
    if (K < jb.Kdim && Nc < jb.Ndim)
      v = jb.trans ? jb.src[Nc * jb.Kdim + K] : jb.src[K * jb.Ndim + Nc];
    dst[idx] = (_Float16)v;
  }
}

// ---- Kernel 1: edge MLP forward + per-node message reduction ---------------
__global__ __launch_bounds__(256) void edge_fwd_kernel(
    const float* __restrict__ q, const float* __restrict__ p,
    const float* __restrict__ be1, const float* __restrict__ be2,
    const float* __restrict__ be3, const _Float16* __restrict__ swz,
    float* __restrict__ msg) {
  __shared__ _Float16 sFeat[64 * 32];
  __shared__ _Float16 sH1[64 * 128];
  __shared__ _Float16 sH2[64 * 128];
  __shared__ float sMsg[32];
  const int tid = threadIdx.x, i = blockIdx.x;
  const int wave = __builtin_amdgcn_readfirstlane(tid >> 5);  // wave-uniform
  const float qi0 = q[i * 3], qi1 = q[i * 3 + 1], qi2 = q[i * 3 + 2];
  const float pi0 = p[i * 3], pi1 = p[i * 3 + 1], pi2 = p[i * 3 + 2];
  for (int k = tid; k < 64 * 32; k += 256) sFeat[k] = (_Float16)0.f;
  if (tid < 32) sMsg[tid] = 0.f;
  __syncthreads();
  const _Float16* We1s = swz + T_We1 * 512;
  const _Float16* We2s = swz + T_We2 * 512;
  const _Float16* We3s = swz + T_We3 * 512;
  for (int t = 0; t < 16; ++t) {
    if (t < 15) __builtin_prefetch(&q[(t + 1) * 64 * 3], 0, 0);
    if (tid < 64) {
      int j = t * 64 + tid;
      float u0 = q[j * 3] - qi0, u1 = q[j * 3 + 1] - qi1, u2 = q[j * 3 + 2] - qi2;
      float pj0 = p[j * 3], pj1 = p[j * 3 + 1], pj2 = p[j * 3 + 2];
      float nrm = fast_sqrt(u0 * u0 + u1 * u1 + u2 * u2);
      float dist = nrm + 1e-9f, inv = fast_rcp(dist);
      float f0 = dist, f1 = inv;
      float f2 = pi0 * pj0 + pi1 * pj1 + pi2 * pj2;
      float f3 = pi0 * u0 + pi1 * u1 + pi2 * u2;
      float f4 = pj0 * u0 + pj1 * u1 + pj2 * u2;
      float f5 = dist * inv;
      if (j == i) { f0 = f1 = f2 = f3 = f4 = f5 = 0.f; }
      _Float16* fr = &sFeat[tid * 32];
      fr[0] = (_Float16)f0; fr[1] = (_Float16)f1; fr[2] = (_Float16)f2;
      fr[3] = (_Float16)f3; fr[4] = (_Float16)f4; fr[5] = (_Float16)f5;
    }
    __syncthreads();
#pragma unroll
    for (int k = 0; k < 4; ++k) {  // 32 tiles, 4 per wave
      int ot = wave + k * 8, mt = ot >> 3, nt = ot & 7;
      v8f acc = wmma_mm<1>(sFeat, 32, mt, We1s, 8, nt);
      epi_silu<false>(acc, be1, sH1, nullptr, 128, mt, nt);
    }
    __syncthreads();
#pragma unroll
    for (int k = 0; k < 4; ++k) {
      int ot = wave + k * 8, mt = ot >> 3, nt = ot & 7;
      v8f acc = wmma_mm<4>(sH1, 128, mt, We2s, 8, nt);
      epi_silu<false>(acc, be2, sH2, nullptr, 128, mt, nt);
    }
    __syncthreads();
    {  // 8 tiles, 1 per wave
      int ot = wave, mt = ot >> 1, nt = ot & 1;
      v8f acc = wmma_mm<4>(sH2, 128, mt, We3s, 2, nt);
      int l = tid & 31, col = nt * 16 + (l & 15), row0 = mt * 16 + ((l >> 4) << 3);
      float b = be3[col], part = 0.f;
#pragma unroll
      for (int r = 0; r < 8; ++r) {
        int j = t * 64 + row0 + r;
        if (j != i) part += acc[r] + b;
      }
      atomicAdd(&sMsg[col], part);
    }
    __syncthreads();
  }
  if (tid < 32) msg[i * 32 + tid] = sMsg[tid];
}

// ---- Kernel 2: node + readout MLP fwd & bwd; H and dm ----------------------
__global__ __launch_bounds__(256) void node_kernel(
    const float* __restrict__ msg, const float* __restrict__ bn1,
    const float* __restrict__ bn2, const float* __restrict__ bn3,
    const float* __restrict__ br1, const float* __restrict__ br2,
    const float* __restrict__ Wr3, const float* __restrict__ br3,
    const _Float16* __restrict__ swz, float* __restrict__ dm,
    float* __restrict__ out) {
  __shared__ _Float16 sM[32 * 32];
  __shared__ _Float16 sHn[32 * 32];
  __shared__ _Float16 sA0[32 * 128];
  __shared__ _Float16 sA1[32 * 128];
  __shared__ _Float16 sPn1[32 * 128];
  __shared__ _Float16 sPn2[32 * 128];
  __shared__ _Float16 sPr1[32 * 128];
  __shared__ _Float16 sPr2[32 * 128];
  __shared__ float sS[32];
  const int tid = threadIdx.x;
  const int wave = __builtin_amdgcn_readfirstlane(tid >> 5);
  const int n0 = blockIdx.x * 32;
  for (int k = tid; k < 32 * 32; k += 256) {
    sM[k] = (_Float16)msg[n0 * 32 + k];
    sHn[k] = (_Float16)0.f;
  }
  __syncthreads();
  // forward: node MLP (16 tiles -> 2 per wave)
#pragma unroll
  for (int k = 0; k < 2; ++k) {
    int ot = wave + k * 8, mt = ot >> 3, nt = ot & 7;
    v8f acc = wmma_mm<1>(sM, 32, mt, swz + T_Wn1 * 512, 8, nt);
    epi_silu<true>(acc, bn1, sA0, sPn1, 128, mt, nt);
  }
  __syncthreads();
#pragma unroll
  for (int k = 0; k < 2; ++k) {
    int ot = wave + k * 8, mt = ot >> 3, nt = ot & 7;
    v8f acc = wmma_mm<4>(sA0, 128, mt, swz + T_Wn2 * 512, 8, nt);
    epi_silu<true>(acc, bn2, sA1, sPn2, 128, mt, nt);
  }
  __syncthreads();
  if (wave < 2) {
    v8f acc = wmma_mm<4>(sA1, 128, wave, swz + T_Wn3 * 512, 1, 0);
    epi_lin_f16(acc, bn3, sHn, 32, wave, 0);
  }
  __syncthreads();
  // forward: readout MLP
#pragma unroll
  for (int k = 0; k < 2; ++k) {
    int ot = wave + k * 8, mt = ot >> 3, nt = ot & 7;
    v8f acc = wmma_mm<1>(sHn, 32, mt, swz + T_Wr1 * 512, 8, nt);
    epi_silu<true>(acc, br1, sA0, sPr1, 128, mt, nt);
  }
  __syncthreads();
#pragma unroll
  for (int k = 0; k < 2; ++k) {
    int ot = wave + k * 8, mt = ot >> 3, nt = ot & 7;
    v8f acc = wmma_mm<4>(sA0, 128, mt, swz + T_Wr2 * 512, 8, nt);
    epi_silu<true>(acc, br2, sA1, sPr2, 128, mt, nt);
  }
  __syncthreads();
  if (tid < 32) {
    float s = br3[0];
    for (int k = 0; k < 128; ++k) s += (float)sA1[tid * 128 + k] * Wr3[k];
    sS[tid] = s;
  }
  __syncthreads();
  if (tid == 0) {
    float h = 0.f;
    for (int k = 0; k < 32; ++k) h += sS[k];
    atomicAdd(out, h);
  }
  // backward: dy2 = Wr3 * silu'(y2)
  for (int k = tid; k < 32 * 128; k += 256)
    sA0[k] = (_Float16)(Wr3[k & 127] * (float)sPr2[k]);
  __syncthreads();
#pragma unroll
  for (int k = 0; k < 2; ++k) {
    int ot = wave + k * 8, mt = ot >> 3, nt = ot & 7;
    v8f acc = wmma_mm<4>(sA0, 128, mt, swz + T_Wr2T * 512, 8, nt);
    epi_mulsp(acc, sPr1, sA1, 128, mt, nt);   // dy1
  }
  __syncthreads();
  if (wave < 2) {
    v8f acc = wmma_mm<4>(sA1, 128, wave, swz + T_Wr1T * 512, 1, 0);
    epi_lin_f16(acc, nullptr, sHn, 32, wave, 0);  // dhn
  }
  __syncthreads();
#pragma unroll
  for (int k = 0; k < 2; ++k) {
    int ot = wave + k * 8, mt = ot >> 3, nt = ot & 7;
    v8f acc = wmma_mm<1>(sHn, 32, mt, swz + T_Wn3T * 512, 8, nt);
    epi_mulsp(acc, sPn2, sA0, 128, mt, nt);   // dz2
  }
  __syncthreads();
#pragma unroll
  for (int k = 0; k < 2; ++k) {
    int ot = wave + k * 8, mt = ot >> 3, nt = ot & 7;
    v8f acc = wmma_mm<4>(sA0, 128, mt, swz + T_Wn2T * 512, 8, nt);
    epi_mulsp(acc, sPn1, sA1, 128, mt, nt);   // dz1
  }
  __syncthreads();
  if (wave < 4) {
    int mt = wave >> 1, nt = wave & 1;
    v8f acc = wmma_mm<4>(sA1, 128, mt, swz + T_Wn1T * 512, 2, nt);
    int l = tid & 31, col = nt * 16 + (l & 15), row0 = mt * 16 + ((l >> 4) << 3);
#pragma unroll
    for (int r = 0; r < 8; ++r) dm[(n0 + row0 + r) * 32 + col] = acc[r];
  }
}

// ---- Kernel 3: edge backward (recompute fwd, backprop, q/p grads) ----------
__global__ __launch_bounds__(256) void edge_bwd_kernel(
    const float* __restrict__ q, const float* __restrict__ p,
    const float* __restrict__ be1, const float* __restrict__ be2,
    const float* __restrict__ We3, const float* __restrict__ dm,
    const _Float16* __restrict__ swz, float* __restrict__ out) {
  __shared__ _Float16 sFeat[64 * 32];
  __shared__ _Float16 sA1[64 * 128];   // a1, later dz1
  __shared__ _Float16 sP1[64 * 128];   // silu'(z1)
  __shared__ _Float16 sDz[64 * 128];   // dz2
  __shared__ float sDf[64 * 16];       // dfeat (f32)
  __shared__ float sG[32];
  __shared__ float sDh[128];
  __shared__ float sAcc[6];
  const int tid = threadIdx.x, i = blockIdx.x;
  const int wave = __builtin_amdgcn_readfirstlane(tid >> 5);
  if (tid < 32) sG[tid] = dm[i * 32 + tid];
  if (tid < 6) sAcc[tid] = 0.f;
  for (int k = tid; k < 64 * 32; k += 256) sFeat[k] = (_Float16)0.f;
  __syncthreads();
  if (tid < 128) {  // dh2c = dm[i] @ We3^T  (same for every edge of this block)
    float s = 0.f;
    for (int c = 0; c < 32; ++c) s += sG[c] * We3[tid * 32 + c];
    sDh[tid] = s;
  }
  const float qi0 = q[i * 3], qi1 = q[i * 3 + 1], qi2 = q[i * 3 + 2];
  const float pi0 = p[i * 3], pi1 = p[i * 3 + 1], pi2 = p[i * 3 + 2];
  float aq0 = 0.f, aq1 = 0.f, aq2 = 0.f, ap0 = 0.f, ap1 = 0.f, ap2 = 0.f;
  float u0 = 0.f, u1 = 0.f, u2 = 0.f, pj0 = 0.f, pj1 = 0.f, pj2 = 0.f;
  float nrm = 1.f, dist = 1.f, inv = 1.f;
  __syncthreads();
  for (int t = 0; t < 16; ++t) {
    if (tid < 64) {
      int j = t * 64 + tid;
      u0 = q[j * 3] - qi0; u1 = q[j * 3 + 1] - qi1; u2 = q[j * 3 + 2] - qi2;
      pj0 = p[j * 3]; pj1 = p[j * 3 + 1]; pj2 = p[j * 3 + 2];
      nrm = fast_sqrt(u0 * u0 + u1 * u1 + u2 * u2);
      dist = nrm + 1e-9f; inv = fast_rcp(dist);
      float f0 = dist, f1 = inv;
      float f2 = pi0 * pj0 + pi1 * pj1 + pi2 * pj2;
      float f3 = pi0 * u0 + pi1 * u1 + pi2 * u2;
      float f4 = pj0 * u0 + pj1 * u1 + pj2 * u2;
      float f5 = dist * inv;
      if (j == i) { f0 = f1 = f2 = f3 = f4 = f5 = 0.f; }
      _Float16* fr = &sFeat[tid * 32];
      fr[0] = (_Float16)f0; fr[1] = (_Float16)f1; fr[2] = (_Float16)f2;
      fr[3] = (_Float16)f3; fr[4] = (_Float16)f4; fr[5] = (_Float16)f5;
    }
    __syncthreads();
#pragma unroll
    for (int k = 0; k < 4; ++k) {  // z1 -> a1, silu'
      int ot = wave + k * 8, mt = ot >> 3, nt = ot & 7;
      v8f acc = wmma_mm<1>(sFeat, 32, mt, swz + T_We1 * 512, 8, nt);
      epi_silu<true>(acc, be1, sA1, sP1, 128, mt, nt);
    }
    __syncthreads();
#pragma unroll
    for (int k = 0; k < 4; ++k) {  // dz2 = dh2c * silu'(z2) directly
      int ot = wave + k * 8, mt = ot >> 3, nt = ot & 7;
      v8f acc = wmma_mm<4>(sA1, 128, mt, swz + T_We2 * 512, 8, nt);
      int l = tid & 31, col = nt * 16 + (l & 15), row0 = mt * 16 + ((l >> 4) << 3);
      float b = be2[col], dh = sDh[col];
#pragma unroll
      for (int r = 0; r < 8; ++r) {
        float z = acc[r] + b;
        float s = fast_sig(z);
        sDz[(row0 + r) * 128 + col] = (_Float16)(dh * (s + z * s * (1.f - s)));
      }
    }
    __syncthreads();
#pragma unroll
    for (int k = 0; k < 4; ++k) {  // dz1 = (dz2 @ W2^T) * silu'(z1)
      int ot = wave + k * 8, mt = ot >> 3, nt = ot & 7;
      v8f acc = wmma_mm<4>(sDz, 128, mt, swz + T_We2T * 512, 8, nt);
      epi_mulsp(acc, sP1, sA1, 128, mt, nt);
    }
    __syncthreads();
    if (wave < 4) {  // dfeat = dz1 @ W1^T (N padded to 16)
      v8f acc = wmma_mm<4>(sA1, 128, wave, swz + T_We1T * 512, 1, 0);
      int l = tid & 31, col = (l & 15), row0 = wave * 16 + ((l >> 4) << 3);
#pragma unroll
      for (int r = 0; r < 8; ++r) sDf[(row0 + r) * 16 + col] = acc[r];
    }
    __syncthreads();
    if (tid < 64) {
      int j = t * 64 + tid;
      if (j != i) {
        const float* g = &sDf[tid * 16];
        float g0 = g[0], g1 = g[1], g2 = g[2], g3 = g[3], g4 = g[4], g5 = g[5];
        float Gd = g0 - g1 * inv * inv + g5 * (inv - dist * inv * inv);
        float cu = Gd * fast_rcp(nrm);
        float gqj0 = cu * u0 + g3 * pi0 + g4 * pj0;
        float gqj1 = cu * u1 + g3 * pi1 + g4 * pj1;
        float gqj2 = cu * u2 + g3 * pi2 + g4 * pj2;
        // node-i accumulators (gq_i = -gq_j)
        aq0 -= gqj0; aq1 -= gqj1; aq2 -= gqj2;
        ap0 += g2 * pj0 + g3 * u0;
        ap1 += g2 * pj1 + g3 * u1;
        ap2 += g2 * pj2 + g3 * u2;
        // node-j contributions: dq_dt = gp, dp_dt = -gq
        atomicAdd(&out[OUT_DQ + j * 3 + 0], g2 * pi0 + g4 * u0);
        atomicAdd(&out[OUT_DQ + j * 3 + 1], g2 * pi1 + g4 * u1);
        atomicAdd(&out[OUT_DQ + j * 3 + 2], g2 * pi2 + g4 * u2);
        atomicAdd(&out[OUT_DP + j * 3 + 0], -gqj0);
        atomicAdd(&out[OUT_DP + j * 3 + 1], -gqj1);
        atomicAdd(&out[OUT_DP + j * 3 + 2], -gqj2);
      }
    }
    __syncthreads();
  }
  if (tid < 64) {
    atomicAdd(&sAcc[0], aq0); atomicAdd(&sAcc[1], aq1); atomicAdd(&sAcc[2], aq2);
    atomicAdd(&sAcc[3], ap0); atomicAdd(&sAcc[4], ap1); atomicAdd(&sAcc[5], ap2);
  }
  __syncthreads();
  if (tid < 3) atomicAdd(&out[OUT_DQ + i * 3 + tid], sAcc[3 + tid]);
  if (tid >= 3 && tid < 6) atomicAdd(&out[OUT_DP + i * 3 + (tid - 3)], -sAcc[tid - 3]);
}

// ---------------------------------------------------------------------------
extern "C" void kernel_launch(void* const* d_in, const int* in_sizes, int n_in,
                              void* d_out, int out_size, void* d_ws, size_t ws_size,
                              hipStream_t stream) {
  (void)in_sizes; (void)n_in; (void)ws_size;
  const float* q   = (const float*)d_in[0];
  const float* p   = (const float*)d_in[1];
  const float* We1 = (const float*)d_in[2];  const float* be1 = (const float*)d_in[3];
  const float* We2 = (const float*)d_in[4];  const float* be2 = (const float*)d_in[5];
  const float* We3 = (const float*)d_in[6];  const float* be3 = (const float*)d_in[7];
  const float* Wn1 = (const float*)d_in[8];  const float* bn1 = (const float*)d_in[9];
  const float* Wn2 = (const float*)d_in[10]; const float* bn2 = (const float*)d_in[11];
  const float* Wn3 = (const float*)d_in[12]; const float* bn3 = (const float*)d_in[13];
  const float* Wr1 = (const float*)d_in[14]; const float* br1 = (const float*)d_in[15];
  const float* Wr2 = (const float*)d_in[16]; const float* br2 = (const float*)d_in[17];
  const float* Wr3 = (const float*)d_in[18]; const float* br3 = (const float*)d_in[19];

  _Float16* swz = (_Float16*)d_ws;
  float* msg = (float*)((char*)d_ws + WS_MSG_BYTES);
  float* dmg = (float*)((char*)d_ws + WS_DM_BYTES);
  float* out = (float*)d_out;

  JobTab jt = {{
      {We1, 6, 128, 8, 0, T_We1},   {We2, 128, 128, 8, 0, T_We2},
      {We3, 128, 32, 2, 0, T_We3},  {We2, 128, 128, 8, 1, T_We2T},
      {We1, 128, 6, 1, 1, T_We1T},  {Wn1, 32, 128, 8, 0, T_Wn1},
      {Wn2, 128, 128, 8, 0, T_Wn2}, {Wn3, 128, 16, 1, 0, T_Wn3},
      {Wr1, 16, 128, 8, 0, T_Wr1},  {Wr2, 128, 128, 8, 0, T_Wr2},
      {Wr2, 128, 128, 8, 1, T_Wr2T},{Wr1, 128, 16, 1, 1, T_Wr1T},
      {Wn3, 16, 128, 8, 1, T_Wn3T}, {Wn2, 128, 128, 8, 1, T_Wn2T},
      {Wn1, 128, 32, 2, 1, T_Wn1T},
  }};

  prep_kernel<<<dim3(16), dim3(256), 0, stream>>>(jt, swz, out, out_size);
  edge_fwd_kernel<<<dim3(NNODES), dim3(256), 0, stream>>>(q, p, be1, be2, be3, swz, msg);
  node_kernel<<<dim3(NNODES / 32), dim3(256), 0, stream>>>(msg, bn1, bn2, bn3, br1, br2,
                                                           Wr3, br3, swz, dmg, out);
  edge_bwd_kernel<<<dim3(NNODES), dim3(256), 0, stream>>>(q, p, be1, be2, We3, dmg, swz,
                                                          out);
}
